// Block_25881472926024
// MI455X (gfx1250) — matrix operand order repaired and checked by reading, not compiled, not run on hardware
//
#include <hip/hip_runtime.h>
#include <hip/hip_bf16.h>

// ---------------------------------------------------------------------------
// Types for CDNA5 WMMA (gfx1250, wave32)
// ---------------------------------------------------------------------------
typedef __bf16 bf16_t;
typedef __attribute__((ext_vector_type(2)))  bf16_t v2bf;
typedef __attribute__((ext_vector_type(8)))  bf16_t v8bf;
typedef __attribute__((ext_vector_type(16))) bf16_t v16bf;
typedef __attribute__((ext_vector_type(8)))  float  v8f;
typedef __attribute__((ext_vector_type(2)))  unsigned int u32x2;
typedef __attribute__((ext_vector_type(4)))  unsigned int u32x4;

// Pack two floats into one dword of bf16 (RNE fptrunc; backend picks best inst)
static __device__ __forceinline__ unsigned int pack2(float a, float b) {
  union { v2bf v; unsigned int u; } cv;
  cv.v[0] = (bf16_t)a;
  cv.v[1] = (bf16_t)b;
  return cv.u;
}

// ---------------------------------------------------------------------------
// bf16-WMMA GEMM:  C[M,N] = epi( A[M,K] @ B[K,N] ), fp32 in/out.
// Tile BM=128, BN=128 (JT*16 used), BK=32; 256 threads = 8 wave32 waves.
// Wave w computes rows [16w,16w+16) x JT*16 cols -> JT WMMA accumulators.
// LDS double-buffered; staging pipelined global->regs / wmma / regs->LDS.
// JT is compile time (8 for full 128-col tiles, 4 for N=64) so the inner
// loop is branch-free: one ds_load clause + JT back-to-back v_wmma.
// Requires M%128==0, K%32==0, and grid.y*JT*16 == N exactly.
// ---------------------------------------------------------------------------
#define BM 128
#define BN 128
#define BK 32

#define EPI_STORE     0
#define EPI_ADD_RES   1   // C = res + A@B
#define EPI_SCALE_ACC 2   // C += rowscale[m*rstride] * (A@B)
#define EPI_SILU_MUL  3   // C = silu(A@B) * res

template <int JT, int EPI>
__global__ __launch_bounds__(256)
void gemm_bf16_wmma(const float* __restrict__ A, const float* __restrict__ B,
                    float* __restrict__ C, const float* __restrict__ res,
                    const float* __restrict__ rowscale, int rstride,
                    int M, int N, int K)
{
  __shared__ unsigned short As[2][BM * BK];   // row-major, stride BK
  __shared__ unsigned short Bs[2][BN * BK];   // TRANSPOSED Bs[n][k], stride BK

  const int tid  = threadIdx.x;
  const int wave = tid >> 5;
  const int lane = tid & 31;
  const int bm   = blockIdx.x * BM;
  const int bn   = blockIdx.y * (JT * 16);

  v8f acc[JT];
#pragma unroll
  for (int j = 0; j < JT; ++j)
#pragma unroll
    for (int i = 0; i < 8; ++i) acc[j][i] = 0.0f;

  // staging assignment: A -> 1 row x 16 cols; B -> 4 k-rows x 4 n-cols
  const int arow = tid >> 1;            // 0..127
  const int acol = (tid & 1) * 16;      // 0 or 16
  const int bng  = (tid & 31) * 4;      // 0..124
  const int bkg  = (tid >> 5) * 4;      // 0..28
  const bool bact = bng < (JT * 16);    // threads staging live B columns

  // fragment geometry (ISA 7.12.2 lane layouts)
  const int r   = lane & 15;
  const int ahi = (lane >> 4) * 8;      // A frag: K-interleave at 8
  const int bhi = (lane >> 4) * 8;      // B frag: K-interleave at 8

  float areg[16];
  float breg[16];                       // breg[kk*4+n]

  auto gload = [&](int k0) {
    const float* ag = A + (size_t)(bm + arow) * K + k0 + acol;
#pragma unroll
    for (int i = 0; i < 16; ++i) areg[i] = ag[i];
    if (bact) {
      const float* bg = B + (size_t)(k0 + bkg) * N + bn + bng;
#pragma unroll
      for (int kk = 0; kk < 4; ++kk)
#pragma unroll
        for (int n = 0; n < 4; ++n) breg[kk * 4 + n] = bg[(size_t)kk * N + n];
    }
  };

  auto lstore = [&](int buf) {
    u32x4 lo = { pack2(areg[0], areg[1]),  pack2(areg[2], areg[3]),
                 pack2(areg[4], areg[5]),  pack2(areg[6], areg[7]) };
    u32x4 hi = { pack2(areg[8], areg[9]),  pack2(areg[10], areg[11]),
                 pack2(areg[12], areg[13]), pack2(areg[14], areg[15]) };
    *(u32x4*)&As[buf][arow * BK + acol]     = lo;
    *(u32x4*)&As[buf][arow * BK + acol + 8] = hi;
    if (bact) {
#pragma unroll
      for (int n = 0; n < 4; ++n) {
        u32x2 p = { pack2(breg[0 + n], breg[4 + n]),
                    pack2(breg[8 + n], breg[12 + n]) };
        *(u32x2*)&Bs[buf][(bng + n) * BK + bkg] = p;
      }
    }
  };

  gload(0);
  lstore(0);
  __syncthreads();

  int buf = 0;
  for (int k0 = 0; k0 < K; k0 += BK) {
    const bool has_next = (k0 + BK) < K;
    if (has_next) gload(k0 + BK);          // global loads fly under the WMMAs

    const unsigned short* ap = &As[buf][(wave * 16 + r) * BK];
    v8bf a_lo = *(const v8bf*)(ap + ahi);
    v8bf a_hi = *(const v8bf*)(ap + 16 + ahi);
    v16bf afrag = __builtin_shufflevector(a_lo, a_hi,
        0,1,2,3,4,5,6,7,8,9,10,11,12,13,14,15);

#pragma unroll
    for (int j = 0; j < JT; ++j) {         // fully unrolled, branch-free
      const unsigned short* bp = &Bs[buf][(j * 16 + r) * BK + bhi];
      v8bf b_lo = *(const v8bf*)(bp);
      v8bf b_hi = *(const v8bf*)(bp + 16);
      v16bf bfrag = __builtin_shufflevector(b_lo, b_hi,
          0,1,2,3,4,5,6,7,8,9,10,11,12,13,14,15);
      acc[j] = __builtin_amdgcn_wmma_f32_16x16x32_bf16(
          false, afrag, false, bfrag, (short)0, acc[j], false, false);
    }

    if (has_next) {
      lstore(buf ^ 1);
      if (k0 + 2 * BK < K) {                // global_prefetch_b8 hints
        __builtin_prefetch(A + (size_t)(bm + arow) * K + k0 + 2 * BK + acol, 0, 1);
        __builtin_prefetch(B + (size_t)(k0 + 2 * BK + bkg) * N + bn + bng, 0, 1);
      }
    }
    __syncthreads();
    buf ^= 1;
  }

  // ---- epilogue (C/D layout: vgpr i -> M = 8*(lane>>4)+i, N = lane&15)
  const int mrow0 = bm + wave * 16 + (lane >> 4) * 8;
#pragma unroll
  for (int j = 0; j < JT; ++j) {
    const int n = bn + j * 16 + r;
#pragma unroll
    for (int i = 0; i < 8; ++i) {
      const int m = mrow0 + i;
      const size_t idx = (size_t)m * N + n;
      float d = acc[j][i];
      if constexpr (EPI == EPI_ADD_RES) {
        d += res[idx];
      } else if constexpr (EPI == EPI_SCALE_ACC) {
        d = C[idx] + rowscale[(size_t)m * rstride] * d;
      } else if constexpr (EPI == EPI_SILU_MUL) {
        d = (d / (1.0f + __expf(-d))) * res[idx];
      }
      C[idx] = d;
    }
  }
}

// ---------------------------------------------------------------------------
// RMSNorm: one block (256 thr) per token
// ---------------------------------------------------------------------------
__global__ __launch_bounds__(256)
void rmsnorm_kernel(const float* __restrict__ x, const float* __restrict__ w,
                    float* __restrict__ out, int Dd)
{
  __shared__ float red[8];
  __shared__ float rs_sh;
  const int t = blockIdx.x;
  const float* xr = x + (size_t)t * Dd;
  float ss = 0.0f;
  for (int i = threadIdx.x; i < Dd; i += 256) { float v = xr[i]; ss += v * v; }
#pragma unroll
  for (int off = 16; off > 0; off >>= 1) ss += __shfl_down(ss, off, 32);
  if ((threadIdx.x & 31) == 0) red[threadIdx.x >> 5] = ss;
  __syncthreads();
  if (threadIdx.x == 0) {
    float s = 0.0f;
#pragma unroll
    for (int i = 0; i < 8; ++i) s += red[i];
    rs_sh = rsqrtf(s / (float)Dd + 1e-6f);
  }
  __syncthreads();
  const float rs = rs_sh;
  for (int i = threadIdx.x; i < Dd; i += 256)
    out[(size_t)t * Dd + i] = xr[i] * rs * w[i];
}

// ---------------------------------------------------------------------------
// RoPE in place on [T, H*64]; pair (i, i+32) within each head
// ---------------------------------------------------------------------------
__global__ __launch_bounds__(256)
void rope_kernel(float* __restrict__ q, const int* __restrict__ start_pos,
                 int S, int H, int T)
{
  const int idx = blockIdx.x * blockDim.x + threadIdx.x;
  if (idx >= T * H * 32) return;
  const int i = idx & 31;
  const int h = (idx >> 5) % H;
  const int t = idx / (32 * H);
  const int s = t % S;
  const float pos  = (float)(s + start_pos[0]);
  const float freq = __powf(10000.0f, -(float)(2 * i) / 64.0f);
  const float ang  = pos * freq;
  const float cs = __cosf(ang), sn = __sinf(ang);
  float* p = q + (size_t)t * (H * 64) + h * 64 + i;
  const float x1 = p[0], x2 = p[32];
  p[0]  = x1 * cs - x2 * sn;
  p[32] = x1 * sn + x2 * cs;
}

// ---------------------------------------------------------------------------
// Causal attention, online softmax. Block = 64 threads = one q-row each.
// Grid = (S/64, H, B). q,k,v,out are [T, H*64] with T = B*S.
// ---------------------------------------------------------------------------
__global__ __launch_bounds__(64)
void attention_kernel(const float* __restrict__ q, const float* __restrict__ k,
                      const float* __restrict__ v, float* __restrict__ out,
                      int B, int S, int H)
{
  __shared__ float Ks[64][65];
  __shared__ float Vs[64][65];
  const int b  = blockIdx.z;
  const int h  = blockIdx.y;
  const int q0 = blockIdx.x * 64;
  const int tq = threadIdx.x;
  const int qi = q0 + tq;
  const int HD = H * 64;

  float qreg[64], accv[64];
  const float* qp = q + (size_t)(b * S + qi) * HD + h * 64;
#pragma unroll
  for (int d = 0; d < 64; ++d) { qreg[d] = qp[d]; accv[d] = 0.0f; }
  float mrun = -1.0e30f, lrun = 0.0f;

  for (int k0 = 0; k0 < q0 + 64; k0 += 64) {
    const float* kp = k + (size_t)(b * S + k0 + tq) * HD + h * 64;
    const float* vp = v + (size_t)(b * S + k0 + tq) * HD + h * 64;
#pragma unroll
    for (int d = 0; d < 64; ++d) { Ks[tq][d] = kp[d]; Vs[tq][d] = vp[d]; }
    __syncthreads();

    const int kmax = (qi - k0 < 63) ? (qi - k0) : 63;
    for (int kk = 0; kk <= kmax; ++kk) {
      float s = 0.0f;
#pragma unroll
      for (int d = 0; d < 64; ++d) s += qreg[d] * Ks[kk][d];
      s *= 0.125f;                     // 1/sqrt(64)
      const float mn = fmaxf(mrun, s);
      const float sc = __expf(mrun - mn);
      const float p  = __expf(s - mn);
      lrun = lrun * sc + p;
#pragma unroll
      for (int d = 0; d < 64; ++d) accv[d] = accv[d] * sc + p * Vs[kk][d];
      mrun = mn;
    }
    __syncthreads();
  }

  float* op = out + (size_t)(b * S + qi) * HD + h * 64;
  const float inv = 1.0f / lrun;
#pragma unroll
  for (int d = 0; d < 64; ++d) op[d] = accv[d] * inv;
}

// ---------------------------------------------------------------------------
// Gating: logits -> softmax -> top2 -> normalized combine + aux accumulators
// ---------------------------------------------------------------------------
__global__ __launch_bounds__(256)
void gate_kernel(const float* __restrict__ h2, const float* __restrict__ gw,
                 float* __restrict__ comb, float* __restrict__ cnt,
                 float* __restrict__ sumP, int T, int Dd)
{
  const int t = blockIdx.x * blockDim.x + threadIdx.x;
  if (t >= T) return;
  float lg[8];
#pragma unroll
  for (int e = 0; e < 8; ++e) lg[e] = 0.0f;
  const float* hr = h2 + (size_t)t * Dd;
  for (int i = 0; i < Dd; ++i) {
    const float hv = hr[i];
    const float* gr = gw + (size_t)i * 8;
#pragma unroll
    for (int e = 0; e < 8; ++e) lg[e] += hv * gr[e];
  }
  float mx = lg[0];
#pragma unroll
  for (int e = 1; e < 8; ++e) mx = fmaxf(mx, lg[e]);
  float pr[8], se = 0.0f;
#pragma unroll
  for (int e = 0; e < 8; ++e) { pr[e] = __expf(lg[e] - mx); se += pr[e]; }
#pragma unroll
  for (int e = 0; e < 8; ++e) pr[e] /= se;
  int i1 = 0;
#pragma unroll
  for (int e = 1; e < 8; ++e) if (pr[e] > pr[i1]) i1 = e;
  int i2 = (i1 == 0) ? 1 : 0;
#pragma unroll
  for (int e = 0; e < 8; ++e) if (e != i1 && pr[e] > pr[i2]) i2 = e;
  const float s2 = pr[i1] + pr[i2];
#pragma unroll
  for (int e = 0; e < 8; ++e)
    comb[(size_t)t * 8 + e] = (e == i1) ? pr[i1] / s2 : (e == i2) ? pr[i2] / s2 : 0.0f;
  atomicAdd(&cnt[i1], 1.0f);
  atomicAdd(&cnt[i2], 1.0f);
#pragma unroll
  for (int e = 0; e < 8; ++e) atomicAdd(&sumP[e], pr[e]);
}

// ---------------------------------------------------------------------------
// Elementwise helpers
// ---------------------------------------------------------------------------
__global__ __launch_bounds__(256)
void zero_kernel(float* __restrict__ p, int n)
{
  const int i = blockIdx.x * blockDim.x + threadIdx.x;
  if (i < n) p[i] = 0.0f;
}

__global__ __launch_bounds__(256)
void add_kernel(const float* __restrict__ a, const float* __restrict__ b,
                float* __restrict__ out, int n)
{
  const int i = blockIdx.x * blockDim.x + threadIdx.x;
  if (i < n) out[i] = a[i] + b[i];
}

__global__ void aux_kernel(const float* __restrict__ cnt,
                           const float* __restrict__ sumP,
                           float* __restrict__ out, int T)
{
  if (threadIdx.x == 0 && blockIdx.x == 0) {
    float s = 0.0f;
#pragma unroll
    for (int e = 0; e < 8; ++e) s += (cnt[e] / (float)T) * (sumP[e] / (float)T);
    out[0] = 8.0f * s;
  }
}

// ---------------------------------------------------------------------------
// Launch
// ---------------------------------------------------------------------------
extern "C" void kernel_launch(void* const* d_in, const int* in_sizes, int n_in,
                              void* d_out, int out_size, void* d_ws, size_t ws_size,
                              hipStream_t stream)
{
  (void)in_sizes; (void)n_in; (void)out_size; (void)ws_size;
  constexpr int Bb = 2, S = 512, D = 768, H = 12, DL = 64, E = 8, HID = 3072;
  constexpr int T = Bb * S, HD = H * 64;

  const float* x       = (const float*)d_in[0];
  const float* norm1_w = (const float*)d_in[1];
  const float* norm2_w = (const float*)d_in[2];
  const float* wq      = (const float*)d_in[3];
  const float* w_dkv   = (const float*)d_in[4];
  const float* w_uk    = (const float*)d_in[5];
  const float* w_uv    = (const float*)d_in[6];
  const float* wo      = (const float*)d_in[7];
  const float* gate_w  = (const float*)d_in[8];
  const float* w1      = (const float*)d_in[9];
  const float* w2      = (const float*)d_in[10];
  const float* w3      = (const float*)d_in[11];
  const int*   sp      = (const int*)d_in[12];
  float*       out     = (float*)d_out;

  float* ws = (float*)d_ws;
  size_t o = 0;
  float* h1   = ws + o; o += (size_t)T * D;
  float* qb   = ws + o; o += (size_t)T * HD;
  float* cb   = ws + o; o += (size_t)T * DL;
  float* kb   = ws + o; o += (size_t)T * HD;
  float* vb   = ws + o; o += (size_t)T * HD;
  float* ab   = ws + o; o += (size_t)T * HD;
  float* x1   = ws + o; o += (size_t)T * D;
  float* h2   = ws + o; o += (size_t)T * D;
  float* comb = ws + o; o += (size_t)T * E;
  float* g3   = ws + o; o += (size_t)T * HID;
  float* gg   = ws + o; o += (size_t)T * HID;
  float* yb   = ws + o; o += (size_t)T * D;
  float* cnt  = ws + o; o += E;
  float* sumP = ws + o; o += E;

  // ---- attention path ----
  rmsnorm_kernel<<<T, 256, 0, stream>>>(x, norm1_w, h1, D);
  gemm_bf16_wmma<8, EPI_STORE><<<dim3(T / BM, HD / BN), 256, 0, stream>>>(
      h1, wq, qb, nullptr, nullptr, 0, T, HD, D);
  gemm_bf16_wmma<4, EPI_STORE><<<dim3(T / BM, 1), 256, 0, stream>>>(
      h1, w_dkv, cb, nullptr, nullptr, 0, T, DL, D);
  gemm_bf16_wmma<8, EPI_STORE><<<dim3(T / BM, HD / BN), 256, 0, stream>>>(
      cb, w_uk, kb, nullptr, nullptr, 0, T, HD, DL);
  gemm_bf16_wmma<8, EPI_STORE><<<dim3(T / BM, HD / BN), 256, 0, stream>>>(
      cb, w_uv, vb, nullptr, nullptr, 0, T, HD, DL);

  const int nrope = T * H * 32;
  rope_kernel<<<(nrope + 255) / 256, 256, 0, stream>>>(qb, sp, S, H, T);
  rope_kernel<<<(nrope + 255) / 256, 256, 0, stream>>>(kb, sp, S, H, T);

  attention_kernel<<<dim3(S / 64, H, Bb), 64, 0, stream>>>(qb, kb, vb, ab, Bb, S, H);

  gemm_bf16_wmma<8, EPI_ADD_RES><<<dim3(T / BM, D / BN), 256, 0, stream>>>(
      ab, wo, x1, x, nullptr, 0, T, D, HD);

  // ---- MoE path ----
  rmsnorm_kernel<<<T, 256, 0, stream>>>(x1, norm2_w, h2, D);
  zero_kernel<<<((T * D) + 255) / 256, 256, 0, stream>>>(yb, T * D);
  zero_kernel<<<1, 256, 0, stream>>>(cnt, 2 * E);   // cnt+sumP contiguous
  gate_kernel<<<(T + 255) / 256, 256, 0, stream>>>(h2, gate_w, comb, cnt, sumP, T, D);

  for (int e = 0; e < E; ++e) {
    const float* w1e = w1 + (size_t)e * D * HID;
    const float* w3e = w3 + (size_t)e * D * HID;
    const float* w2e = w2 + (size_t)e * HID * D;
    gemm_bf16_wmma<8, EPI_STORE><<<dim3(T / BM, HID / BN), 256, 0, stream>>>(
        h2, w3e, g3, nullptr, nullptr, 0, T, HID, D);
    gemm_bf16_wmma<8, EPI_SILU_MUL><<<dim3(T / BM, HID / BN), 256, 0, stream>>>(
        h2, w1e, gg, g3, nullptr, 0, T, HID, D);                 // silu fused
    gemm_bf16_wmma<8, EPI_SCALE_ACC><<<dim3(T / BM, D / BN), 256, 0, stream>>>(
        gg, w2e, yb, nullptr, comb + e, E, T, D, HID);
  }

  // ---- outputs: x_out then aux scalar ----
  add_kernel<<<((T * D) + 255) / 256, 256, 0, stream>>>(x1, yb, out, T * D);
  aux_kernel<<<1, 1, 0, stream>>>(cnt, sumP, out + (size_t)T * D, T);
}
